// NeuroMUSX_77068893159477
// MI455X (gfx1250) — compile-verified, matrix operand order checked by reading, not run on hardware
//
#include <hip/hip_runtime.h>
#include <hip/hip_bf16.h>
#include <stdint.h>

// GATv2 forward (4 layers) for MI455X / gfx1250.
// Dense projections: v_wmma_f32_16x16x32_f16, one wave per 16-row strip with
// A-fragment reuse across 8 column tiles (32 WMMAs/wave, unguarded stores).
// Edge softmax/aggregation: gathered vmem + global float atomics (HBM-bound path).

typedef __attribute__((ext_vector_type(16))) _Float16 v16h;
typedef __attribute__((ext_vector_type(8)))  float    v8f;

#define THREADS 256

static inline int cdiv_ll(long long a, long long b) { return (int)((a + b - 1) / b); }

__device__ __forceinline__ float leakyf(float v, float s) { return v > 0.f ? v : s * v; }

// Sign-split float atomic max (lowers to global_atomic_max/min on int bits).
__device__ __forceinline__ void atomicMaxF(float* addr, float v) {
  if (v >= 0.f) atomicMax((int*)addr, __float_as_int(v));
  else          atomicMin((unsigned int*)addr, __float_as_uint(v));
}

// ---------------------------------------------------------------------------
__global__ void fill_f32(float* __restrict__ p, float v, int n) {
  int i = blockIdx.x * blockDim.x + threadIdx.x;
  if (i < n) p[i] = v;
}

// ---------------------------------------------------------------------------
// Layer-0 linear (din = 2): xl = x@Wl + bl, xr = x@Wr + br   (per node, 128 out)
// ---------------------------------------------------------------------------
__global__ void lin_din2(const float* __restrict__ x,
                         const float* __restrict__ Wl, const float* __restrict__ bl,
                         const float* __restrict__ Wr, const float* __restrict__ br,
                         float* __restrict__ xl, float* __restrict__ xr, int N) {
  int t = blockIdx.x * blockDim.x + threadIdx.x;
  if (t >= N * 128) return;
  int i = t >> 7, hc = t & 127;
  float x0 = x[2 * i], x1 = x[2 * i + 1];
  xl[t] = x0 * Wl[hc] + x1 * Wl[128 + hc] + bl[hc];
  xr[t] = x0 * Wr[hc] + x1 * Wr[128 + hc] + br[hc];
}

// ---------------------------------------------------------------------------
// Pack fp32 activations [Nrows,128] into the gfx1250 WMMA 16-bit A-operand
// layout (16x32 per k-step): lane<16 -> M=lane, K base 0; lane>=16 -> K base 8.
// Layout: Apack[ ((tile*4 + s)*32 + lane)*16 + j ].  Rows >= Nrows zero-filled.
// ---------------------------------------------------------------------------
__global__ void pack_a_f16(const float* __restrict__ A, _Float16* __restrict__ Apack,
                           int MT, int Nrows) {
  int id = blockIdx.x * blockDim.x + threadIdx.x;
  if (id >= MT * 2048) return;
  int t = id >> 11;
  int r = id & 2047;
  int s = r >> 9;
  int l = (r >> 4) & 31;
  int j = r & 15;
  int row   = t * 16 + (l & 15);
  int kbase = (l >> 4) * 8;
  int p = j >> 1, hl = j & 1;
  int kk = kbase + ((p < 4) ? (2 * p) : (16 + 2 * (p - 4))) + hl;
  int K  = s * 32 + kk;
  float v = (row < Nrows) ? A[(size_t)row * 128 + K] : 0.f;
  Apack[id] = (_Float16)v;
}

// Pack fp32 weights [128, Nout] (row-major) into the mirrored B-operand layout.
// Layout: Bpack[ ((s*CT + ct)*32 + lane)*16 + j ]
__global__ void pack_b_f16(const float* __restrict__ Bm, _Float16* __restrict__ Bpack,
                           int CT, int Nout) {
  int id = blockIdx.x * blockDim.x + threadIdx.x;
  if (id >= 2048 * CT) return;
  int sct = id >> 9;
  int s = sct / CT, ct = sct % CT;
  int l = (id >> 4) & 31, j = id & 15;
  int col   = ct * 16 + (l & 15);
  int kbase = (l >> 4) * 8;
  int p = j >> 1, hl = j & 1;
  int kk = kbase + ((p < 4) ? (2 * p) : (16 + 2 * (p - 4))) + hl;
  int K  = s * 32 + kk;
  Bpack[id] = (_Float16)Bm[(size_t)K * Nout + col];
}

// ---------------------------------------------------------------------------
// WMMA GEMM strip kernel: out[MT*16, Nout] = A(f16 packed) @ B(f16 packed) + bias.
// One wave per 16-row strip; A fragment loaded once per k-step and reused
// across CT column tiles (CT*4 wmma per wave). Output buffer is padded to
// MT*16 rows, so stores are unguarded; only wave-uniform early-out -> EXEC
// stays all-ones around v_wmma (ISA requirement).
// ---------------------------------------------------------------------------
template <int CT>
__global__ void wmma_gemm_strip(const _Float16* __restrict__ Apack,
                                const _Float16* __restrict__ Bpack,
                                const float* __restrict__ bias,
                                float* __restrict__ out, int MT) {
  const int Nout = CT * 16;
  int wave = (int)((blockIdx.x * blockDim.x + threadIdx.x) >> 5);
  int lane = threadIdx.x & 31;
  if (wave >= MT) return;
  int mt = wave;
  int colb = lane & 15;

  v8f c[CT];
#pragma unroll
  for (int ct = 0; ct < CT; ++ct) {
    float bv = bias[ct * 16 + colb];
#pragma unroll
    for (int v = 0; v < 8; ++v) c[ct][v] = bv;
  }
#pragma unroll
  for (int s = 0; s < 4; ++s) {
    v16h a = *(const v16h*)(Apack + ((size_t)(mt * 4 + s) * 32 + lane) * 16);
#pragma unroll
    for (int ct = 0; ct < CT; ++ct) {
      v16h b = *(const v16h*)(Bpack + ((size_t)(s * CT + ct) * 32 + lane) * 16);
      c[ct] = __builtin_amdgcn_wmma_f32_16x16x32_f16(false, a, false, b, (short)0,
                                                     c[ct], false, false);
    }
  }
  int rbase = mt * 16 + ((lane >> 4) << 3);  // lanes 16..31 hold rows 8..15 of tile
#pragma unroll
  for (int ct = 0; ct < CT; ++ct) {
    int col = ct * 16 + colb;
#pragma unroll
    for (int v = 0; v < 8; ++v)
      out[(size_t)(rbase + v) * Nout + col] = c[ct][v];
  }
}

// ---------------------------------------------------------------------------
// Edge pass A (C=8): logits[e,h] = att_h . leaky( xl[src]+xr[dst]+We.e , 0.2 )
// + atomic segment-max into nmax[dst,h]. One thread per (edge, head).
// ---------------------------------------------------------------------------
__global__ void edge8_logits(const int* __restrict__ ei, const float* __restrict__ ea,
                             const float* __restrict__ xl, const float* __restrict__ xr,
                             const float* __restrict__ We, const float* __restrict__ att,
                             float* __restrict__ logits, float* __restrict__ nmax, int E) {
  int t = blockIdx.x * blockDim.x + threadIdx.x;
  if (t >= E * 16) return;
  int e = t >> 4, h = t & 15;
  __builtin_prefetch(ei + e + 2048, 0, 1);       // streaming index prefetch (global_prefetch_b8)
  int src = ei[e], dst = ei[E + e];
  float a0 = ea[2 * e], a1 = ea[2 * e + 1];
  const float4* pl = (const float4*)(xl + (size_t)src * 128 + h * 8);
  const float4* pr = (const float4*)(xr + (size_t)dst * 128 + h * 8);
  const float4* w0 = (const float4*)(We + h * 8);
  const float4* w1 = (const float4*)(We + 128 + h * 8);
  const float4* pa = (const float4*)(att + h * 8);
  float logit = 0.f;
#pragma unroll
  for (int k = 0; k < 2; ++k) {
    float4 l = pl[k], r = pr[k], u = w0[k], v = w1[k], av = pa[k];
    logit += leakyf(l.x + r.x + a0 * u.x + a1 * v.x, 0.2f) * av.x;
    logit += leakyf(l.y + r.y + a0 * u.y + a1 * v.y, 0.2f) * av.y;
    logit += leakyf(l.z + r.z + a0 * u.z + a1 * v.z, 0.2f) * av.z;
    logit += leakyf(l.w + r.w + a0 * u.w + a1 * v.w, 0.2f) * av.w;
  }
  logits[t] = logit;
  atomicMaxF(&nmax[dst * 16 + h], logit);
}

// Edge pass B (C=8): ex = exp(logit - max); accumulate denom and ex*xl[src].
__global__ void edge8_accum(const int* __restrict__ ei, const float* __restrict__ logits,
                            const float* __restrict__ nmax, const float* __restrict__ xl,
                            float* __restrict__ den, float* __restrict__ acc, int E) {
  int t = blockIdx.x * blockDim.x + threadIdx.x;
  if (t >= E * 16) return;
  int e = t >> 4, h = t & 15;
  __builtin_prefetch(logits + t + 4096, 0, 1);
  int src = ei[e], dst = ei[E + e];
  float ex = __expf(logits[t] - nmax[dst * 16 + h]);
  atomicAdd(&den[dst * 16 + h], ex);
  const float4* pl = (const float4*)(xl + (size_t)src * 128 + h * 8);
  float* pacc = acc + (size_t)dst * 128 + h * 8;
#pragma unroll
  for (int k = 0; k < 2; ++k) {
    float4 l = pl[k];
    atomicAdd(pacc + 4 * k + 0, ex * l.x);
    atomicAdd(pacc + 4 * k + 1, ex * l.y);
    atomicAdd(pacc + 4 * k + 2, ex * l.z);
    atomicAdd(pacc + 4 * k + 3, ex * l.w);
  }
}

// C=1 variants (final / sbs heads)
__global__ void edge1_logits(const int* __restrict__ ei, const float* __restrict__ ea,
                             const float* __restrict__ xl, const float* __restrict__ xr,
                             const float* __restrict__ We, const float* __restrict__ att,
                             float* __restrict__ logits, float* __restrict__ nmax, int E) {
  int t = blockIdx.x * blockDim.x + threadIdx.x;
  if (t >= E * 16) return;
  int e = t >> 4, h = t & 15;
  int src = ei[e], dst = ei[E + e];
  float m = xl[(size_t)src * 16 + h] + xr[(size_t)dst * 16 + h]
          + ea[2 * e] * We[h] + ea[2 * e + 1] * We[16 + h];
  float lg = leakyf(m, 0.2f) * att[h];
  logits[t] = lg;
  atomicMaxF(&nmax[dst * 16 + h], lg);
}

__global__ void edge1_accum(const int* __restrict__ ei, const float* __restrict__ logits,
                            const float* __restrict__ nmax, const float* __restrict__ xl,
                            float* __restrict__ den, float* __restrict__ acc, int E) {
  int t = blockIdx.x * blockDim.x + threadIdx.x;
  if (t >= E * 16) return;
  int e = t >> 4, h = t & 15;
  int src = ei[e], dst = ei[E + e];
  float ex = __expf(logits[t] - nmax[dst * 16 + h]);
  atomicAdd(&den[dst * 16 + h], ex);
  atomicAdd(&acc[(size_t)dst * 16 + h], ex * xl[(size_t)src * 16 + h]);
}

// Finalize concat layer: h_out = leaky( acc/den + bias , 0.01 )
__global__ void finalize_concat(const float* __restrict__ acc, const float* __restrict__ den,
                                const float* __restrict__ bias, float* __restrict__ out, int N) {
  int t = blockIdx.x * blockDim.x + threadIdx.x;
  if (t >= N * 128) return;
  int i = t >> 7, hc = t & 127, h = hc >> 3;
  float d = den[i * 16 + h];
  float v = (d > 0.f) ? (acc[t] / d) : 0.f;
  out[t] = leakyf(v + bias[hc], 0.01f);
}

// Finalize mean-over-heads scalar layer.
// mode 0 (final): out_node[i] = (mean_h + bias) * mask[i]
// mode 1 (sbs)  : atomic accumulate per-graph sum / count
__global__ void finalize_scalar(const float* __restrict__ acc, const float* __restrict__ den,
                                const float* __restrict__ bias, const float* __restrict__ mask,
                                const int* __restrict__ batch, float* __restrict__ out_node,
                                float* __restrict__ satsum, float* __restrict__ satcnt,
                                int N, int mode) {
  int i = blockIdx.x * blockDim.x + threadIdx.x;
  if (i >= N) return;
  float s = 0.f;
#pragma unroll
  for (int h = 0; h < 16; ++h) {
    float d = den[i * 16 + h];
    s += (d > 0.f) ? acc[i * 16 + h] / d : 0.f;
  }
  float v = s * (1.f / 16.f) + bias[0];
  if (mode == 0) {
    out_node[i] = v * mask[i];
  } else {
    atomicAdd(&satsum[batch[i]], v);
    atomicAdd(&satcnt[batch[i]], 1.f);
  }
}

__global__ void satdiv(const float* __restrict__ ss, const float* __restrict__ sc,
                       float* __restrict__ out, int B) {
  int b = blockIdx.x * blockDim.x + threadIdx.x;
  if (b < B) out[b] = ss[b] / fmaxf(sc[b], 1.f);
}

// ---------------------------------------------------------------------------
extern "C" void kernel_launch(void* const* d_in, const int* in_sizes, int n_in,
                              void* d_out, int out_size, void* d_ws, size_t ws_size,
                              hipStream_t stream) {
  const int N = in_sizes[0] / 2;   // x is [N,2]
  const int E = in_sizes[1] / 2;   // edge_index is [2,E]
  const int B = out_size - N;
  const int MT = (N + 15) / 16;
  const int NP = MT * 16;          // row-padded size for GEMM outputs

  const float* x     = (const float*)d_in[0];
  const int*   ei    = (const int*)  d_in[1];
  const float* ea    = (const float*)d_in[2];
  const int*   batch = (const int*)  d_in[3];
  const float* mask  = (const float*)d_in[4];
  // params flattened leaf-order: per layer {Wl, bl, Wr, br, We, att, bias}
  const float *Wl0 = (const float*)d_in[5],  *bl0 = (const float*)d_in[6],
              *Wr0 = (const float*)d_in[7],  *br0 = (const float*)d_in[8],
              *We0 = (const float*)d_in[9],  *att0 = (const float*)d_in[10],
              *bias0 = (const float*)d_in[11];
  const float *Wl1 = (const float*)d_in[12], *bl1 = (const float*)d_in[13],
              *Wr1 = (const float*)d_in[14], *br1 = (const float*)d_in[15],
              *We1 = (const float*)d_in[16], *att1 = (const float*)d_in[17],
              *bias1 = (const float*)d_in[18];
  const float *WlF = (const float*)d_in[19], *blF = (const float*)d_in[20],
              *WrF = (const float*)d_in[21], *brF = (const float*)d_in[22],
              *WeF = (const float*)d_in[23], *attF = (const float*)d_in[24],
              *biasF = (const float*)d_in[25];
  const float *WlS = (const float*)d_in[26], *blS = (const float*)d_in[27],
              *WrS = (const float*)d_in[28], *brS = (const float*)d_in[29],
              *WeS = (const float*)d_in[30], *attS = (const float*)d_in[31],
              *biasS = (const float*)d_in[32];

  // ---- workspace bump allocator (~360 MB total)
  char* base = (char*)d_ws;
  size_t off = 0;
  auto alloc = [&](size_t bytes) -> void* {
    void* p = base + off;
    off = (off + bytes + 255) & ~(size_t)255;
    return p;
  };
  float* h0     = (float*)alloc((size_t)N  * 128 * 4);
  float* h1     = (float*)alloc((size_t)N  * 128 * 4);
  float* xl     = (float*)alloc((size_t)NP * 128 * 4);  // padded: GEMM target
  float* xr     = (float*)alloc((size_t)NP * 128 * 4);  // padded: GEMM target
  float* acc    = (float*)alloc((size_t)N  * 128 * 4);
  float* nmax   = (float*)alloc((size_t)N * 16 * 4);
  float* den    = (float*)alloc((size_t)N * 16 * 4);
  float* logits = (float*)alloc((size_t)E * 16 * 4);
  _Float16* Apack = (_Float16*)alloc((size_t)MT * 2048 * 2);
  _Float16* Bpack = (_Float16*)alloc((size_t)2048 * 8 * 2);
  float* satsum = (float*)alloc((size_t)B * 4);
  float* satcnt = (float*)alloc((size_t)B * 4);

  float* out_node = (float*)d_out;       // x_out [N]
  float* out_sat  = (float*)d_out + N;   // x_out_sat [B]

  dim3 blk(THREADS);
  const int gNH  = cdiv_ll((long long)N * 128, THREADS);
  const int gN16 = cdiv_ll((long long)N * 16, THREADS);
  const int gEH  = cdiv_ll((long long)E * 16, THREADS);
  const int gN   = cdiv_ll(N, THREADS);
  const int gStrip = cdiv_ll((long long)MT * 32, THREADS);  // 1 wave / 16-row strip
  const int gPackA = cdiv_ll((long long)MT * 2048, THREADS);
  const float NEG = -3.402823466e38f;

  // ================= layer 0 (din=2, C=8, concat) =================
  lin_din2<<<gNH, blk, 0, stream>>>(x, Wl0, bl0, Wr0, br0, xl, xr, N);
  hipMemsetAsync(den, 0, (size_t)N * 16 * 4, stream);
  hipMemsetAsync(acc, 0, (size_t)N * 128 * 4, stream);
  fill_f32<<<gN16, blk, 0, stream>>>(nmax, NEG, N * 16);
  edge8_logits<<<gEH, blk, 0, stream>>>(ei, ea, xl, xr, We0, att0, logits, nmax, E);
  edge8_accum <<<gEH, blk, 0, stream>>>(ei, logits, nmax, xl, den, acc, E);
  finalize_concat<<<gNH, blk, 0, stream>>>(acc, den, bias0, h0, N);

  // ================= layer 1 (din=128, C=8, concat): WMMA GEMMs ===
  pack_a_f16<<<gPackA, blk, 0, stream>>>(h0, Apack, MT, N);
  pack_b_f16<<<cdiv_ll(2048 * 8, THREADS), blk, 0, stream>>>(Wl1, Bpack, 8, 128);
  wmma_gemm_strip<8><<<gStrip, blk, 0, stream>>>(Apack, Bpack, bl1, xl, MT);
  pack_b_f16<<<cdiv_ll(2048 * 8, THREADS), blk, 0, stream>>>(Wr1, Bpack, 8, 128);
  wmma_gemm_strip<8><<<gStrip, blk, 0, stream>>>(Apack, Bpack, br1, xr, MT);

  hipMemsetAsync(den, 0, (size_t)N * 16 * 4, stream);
  hipMemsetAsync(acc, 0, (size_t)N * 128 * 4, stream);
  fill_f32<<<gN16, blk, 0, stream>>>(nmax, NEG, N * 16);
  edge8_logits<<<gEH, blk, 0, stream>>>(ei, ea, xl, xr, We1, att1, logits, nmax, E);
  edge8_accum <<<gEH, blk, 0, stream>>>(ei, logits, nmax, xl, den, acc, E);
  finalize_concat<<<gNH, blk, 0, stream>>>(acc, den, bias1, h1, N);

  // h1 feeds both final and sbs heads: pack once.
  pack_a_f16<<<gPackA, blk, 0, stream>>>(h1, Apack, MT, N);

  // ================= final head (C=1, mean) =======================
  pack_b_f16<<<cdiv_ll(2048, THREADS), blk, 0, stream>>>(WlF, Bpack, 1, 16);
  wmma_gemm_strip<1><<<gStrip, blk, 0, stream>>>(Apack, Bpack, blF, xl, MT);
  pack_b_f16<<<cdiv_ll(2048, THREADS), blk, 0, stream>>>(WrF, Bpack, 1, 16);
  wmma_gemm_strip<1><<<gStrip, blk, 0, stream>>>(Apack, Bpack, brF, xr, MT);

  hipMemsetAsync(den, 0, (size_t)N * 16 * 4, stream);
  hipMemsetAsync(acc, 0, (size_t)N * 16 * 4, stream);
  fill_f32<<<gN16, blk, 0, stream>>>(nmax, NEG, N * 16);
  edge1_logits<<<gEH, blk, 0, stream>>>(ei, ea, xl, xr, WeF, attF, logits, nmax, E);
  edge1_accum <<<gEH, blk, 0, stream>>>(ei, logits, nmax, xl, den, acc, E);
  finalize_scalar<<<gN, blk, 0, stream>>>(acc, den, biasF, mask, batch,
                                          out_node, satsum, satcnt, N, /*mode=*/0);

  // ================= sbs head (C=1, mean, per-graph pooled) =======
  pack_b_f16<<<cdiv_ll(2048, THREADS), blk, 0, stream>>>(WlS, Bpack, 1, 16);
  wmma_gemm_strip<1><<<gStrip, blk, 0, stream>>>(Apack, Bpack, blS, xl, MT);
  pack_b_f16<<<cdiv_ll(2048, THREADS), blk, 0, stream>>>(WrS, Bpack, 1, 16);
  wmma_gemm_strip<1><<<gStrip, blk, 0, stream>>>(Apack, Bpack, brS, xr, MT);

  hipMemsetAsync(den, 0, (size_t)N * 16 * 4, stream);
  hipMemsetAsync(acc, 0, (size_t)N * 16 * 4, stream);
  fill_f32<<<gN16, blk, 0, stream>>>(nmax, NEG, N * 16);
  edge1_logits<<<gEH, blk, 0, stream>>>(ei, ea, xl, xr, WeS, attS, logits, nmax, E);
  edge1_accum <<<gEH, blk, 0, stream>>>(ei, logits, nmax, xl, den, acc, E);

  hipMemsetAsync(satsum, 0, (size_t)B * 4, stream);
  hipMemsetAsync(satcnt, 0, (size_t)B * 4, stream);
  finalize_scalar<<<gN, blk, 0, stream>>>(acc, den, biasS, mask, batch,
                                          out_node, satsum, satcnt, N, /*mode=*/1);
  satdiv<<<1, 64, 0, stream>>>(satsum, satcnt, out_sat, B);
}
